// SelfAttention_21371757265315
// MI455X (gfx1250) — compile-verified, matrix operand order checked by reading, not run on hardware
//
#include <hip/hip_runtime.h>

// ---------------------------------------------------------------------------
// Self-attention forward for B=2, L=2048, E=512, H=8, D=64 on gfx1250 (CDNA5).
// All matmuls run on v_wmma_f32_16x16x32_bf16 (bf16 operands, f32 accum).
// Attention is fused flash-style: scores never hit HBM.
// GEMM tiles are double-buffered through LDS with GLOBAL_LOAD_ASYNC_TO_LDS
// (ASYNCcnt): DMA for tile t+1 overlaps the WMMAs on tile t, one barrier
// per K-step.
// ---------------------------------------------------------------------------

typedef __attribute__((ext_vector_type(16))) __bf16 bf16x16;
typedef __attribute__((ext_vector_type(8)))  __bf16 bf16x8;
typedef __attribute__((ext_vector_type(8)))  float  v8f;
typedef __attribute__((ext_vector_type(4)))  int    v4i;

union Frag {
    bf16x16 v;
    bf16x8  h[2];
};

__device__ __forceinline__ v8f wmma_bf16(bf16x16 a, bf16x16 b, v8f c) {
    // (neg_a, A, neg_b, B, c_mod, C, reuse_a, reuse_b)
    return __builtin_amdgcn_wmma_f32_16x16x32_bf16(
        false, a, false, b, (short)0, c, false, false);
}

constexpr int BB  = 2;
constexpr int LL  = 2048;
constexpr int EE  = 512;
constexpr int HH  = 8;
constexpr int DD  = 64;
constexpr int GM  = BB * LL;   // 4096 rows in every projection GEMM

// ---------------------------------------------------------------------------
// Async global->LDS copy (CDNA5 GLOBAL_LOAD_ASYNC_TO_LDS_B128, ASYNCcnt),
// guarded so an older toolchain falls back to the proven synchronous path.
// Builtin expects (v4i addrspace(1)*, v4i addrspace(3)*, imm offset, imm cpol).
// ---------------------------------------------------------------------------
#if __has_builtin(__builtin_amdgcn_global_load_async_to_lds_b128)
#define HAS_ASYNC_LDS 1
typedef v4i __attribute__((address_space(1)))* as1_v4i_ptr;
typedef v4i __attribute__((address_space(3)))* as3_v4i_ptr;
__device__ __forceinline__ void async_copy_b128(void* lds, const void* gptr) {
    __builtin_amdgcn_global_load_async_to_lds_b128(
        (as1_v4i_ptr)(unsigned long long)gptr,
        (as3_v4i_ptr)(unsigned int)(unsigned long long)lds,
        0, 0);
}
__device__ __forceinline__ void async_wait0() {
#if __has_builtin(__builtin_amdgcn_s_wait_asynccnt)
    __builtin_amdgcn_s_wait_asynccnt(0);
#else
    asm volatile("s_wait_asynccnt 0" ::: "memory");
#endif
}
#else
#define HAS_ASYNC_LDS 0
__device__ __forceinline__ void async_copy_b128(void* lds, const void* gptr) {
    *(float4*)lds = *(const float4*)gptr;
}
__device__ __forceinline__ void async_wait0() {}
#endif

// Convert 8 consecutive f32 (16B-aligned) to a bf16x8 half-fragment.
__device__ __forceinline__ bf16x8 cvt8(const float* p) {
    float4 a = *(const float4*)p;
    float4 b = *(const float4*)(p + 4);
    bf16x8 r;
    r[0] = (__bf16)a.x; r[1] = (__bf16)a.y; r[2] = (__bf16)a.z; r[3] = (__bf16)a.w;
    r[4] = (__bf16)b.x; r[5] = (__bf16)b.y; r[6] = (__bf16)b.z; r[7] = (__bf16)b.w;
    return r;
}

// ---------------------------------------------------------------------------
// Generic projection GEMM: dst = A[M,512] @ W[512,512]^T + bias, M = 4096.
// MODE 0: write bf16 into [B,H,L,D]   (Q and K)
// MODE 1: write bf16 into [B,H,D,L]   (V, pre-transposed for attn 2nd GEMM)
// MODE 2: write f32  into [M, 512]    (final output projection)
// Block: 256 threads (8 waves). Block tile 128x128, wave tile 64x32,
// K stepped by 32. Double-buffered async DMA staging: copies for step t+1
// run concurrently with the WMMAs of step t.
// ---------------------------------------------------------------------------
template <int MODE>
__global__ __launch_bounds__(256, 2)
void gemm_proj(const float* __restrict__ A, const float* __restrict__ W,
               const float* __restrict__ bias, void* __restrict__ dstv) {
    constexpr int PADF   = 4;                    // 36-float rows: 16 rows hit
    constexpr int NSTEPS = 512 / 32;             // 16 distinct bank groups
    __shared__ float sAf[2][128][32 + PADF];
    __shared__ float sWf[2][128][32 + PADF];

    const int tid  = threadIdx.x;
    const int wave = tid >> 5;
    const int lane = tid & 31;
    const int half = lane >> 4;                  // 0 | 1 (lane group)
    const int l16  = lane & 15;
    const int m0   = blockIdx.x * 128;
    const int n0   = blockIdx.y * 128;
    const int wm   = wave & 1;                   // 2 wave-rows  x 64
    const int wn   = wave >> 1;                  // 4 wave-cols  x 32

    // Stage one 128x32 f32 tile pair: 1024 16B chunks per matrix, 4/thread.
    auto stage = [&](int buf, int k0) {
        #pragma unroll
        for (int i = 0; i < 4; ++i) {
            int idx = tid + i * 256;             // 0..1023
            int row = idx >> 3;                  // 0..127
            int cg  = (idx & 7) * 4;             // col group of 4 floats
            async_copy_b128(&sAf[buf][row][cg], A + (size_t)(m0 + row) * 512 + k0 + cg);
            async_copy_b128(&sWf[buf][row][cg], W + (size_t)(n0 + row) * 512 + k0 + cg);
        }
    };

    v8f acc[4][2] = {};

    stage(0, 0);                                 // prologue DMA
    for (int step = 0; step < NSTEPS; ++step) {
        const int buf = step & 1;
        async_wait0();                           // my copies for `buf` done
        __syncthreads();                         // everyone's copies done
        if (step + 1 < NSTEPS)
            stage(buf ^ 1, (step + 1) * 32);     // overlaps with WMMAs below

        Frag aF[4], bF[2];
        #pragma unroll
        for (int i = 0; i < 4; ++i) {
            const float* p = &sAf[buf][wm * 64 + i * 16 + l16][half * 8];
            aF[i].h[0] = cvt8(p);
            aF[i].h[1] = cvt8(p + 16);
        }
        #pragma unroll
        for (int j = 0; j < 2; ++j) {
            const float* p = &sWf[buf][wn * 32 + j * 16 + l16][half * 8];
            bF[j].h[0] = cvt8(p);
            bF[j].h[1] = cvt8(p + 16);
        }
        #pragma unroll
        for (int i = 0; i < 4; ++i)
            #pragma unroll
            for (int j = 0; j < 2; ++j)
                acc[i][j] = wmma_bf16(aF[i].v, bF[j].v, acc[i][j]);
    }

    // Epilogue: C-fragment layout — VGPR r holds row (r + 8*half), col l16.
    #pragma unroll
    for (int i = 0; i < 4; ++i) {
        #pragma unroll
        for (int j = 0; j < 2; ++j) {
            int gn = n0 + wn * 32 + j * 16 + l16;
            float bval = bias[gn];
            #pragma unroll
            for (int r = 0; r < 8; ++r) {
                int gm = m0 + wm * 64 + i * 16 + r + half * 8;
                float val = acc[i][j][r] + bval;
                int b = gm >> 11;                // / L
                int l = gm & (LL - 1);
                if (MODE == 0) {                 // [B,H,L,D] bf16
                    int h = gn >> 6, d = gn & 63;
                    ((__bf16*)dstv)[(((size_t)(b * HH + h) * LL + l) << 6) + d] = (__bf16)val;
                } else if (MODE == 1) {          // [B,H,D,L] bf16
                    int h = gn >> 6, d = gn & 63;
                    ((__bf16*)dstv)[((size_t)(b * HH + h) * DD + d) * LL + l] = (__bf16)val;
                } else {                         // [M,512] f32
                    ((float*)dstv)[(size_t)gm * 512 + gn] = val;
                }
            }
        }
    }
}

// ---------------------------------------------------------------------------
// Flash attention: block = 4 waves, owns 64 q-rows of one (b,h).
// Each wave owns 16 q-rows; loop over Lk in chunks of 64.
// ---------------------------------------------------------------------------
__device__ __forceinline__ float redmax16(float v) {
    #pragma unroll
    for (int m = 1; m < 16; m <<= 1) v = fmaxf(v, __shfl_xor(v, m, 32));
    return v;
}
__device__ __forceinline__ float redsum16(float v) {
    #pragma unroll
    for (int m = 1; m < 16; m <<= 1) v += __shfl_xor(v, m, 32);
    return v;
}

__global__ __launch_bounds__(128, 4)
void attn_flash(const __bf16* __restrict__ qb,   // [B,H,L,D]
                const __bf16* __restrict__ kb,   // [B,H,L,D]
                const __bf16* __restrict__ vb,   // [B,H,D,L]
                const int*    __restrict__ mask, // [B,1,L,L]
                float*        __restrict__ Ob) { // [B,L,E]
    constexpr float SCALE = 0.04419417382415922f;   // 1/sqrt(512)
    __shared__ __bf16 sP[4][16][64 + 8];            // per-wave P relayout tile

    const int wave = threadIdx.x >> 5;
    const int lane = threadIdx.x & 31;
    const int half = lane >> 4;
    const int l16  = lane & 15;
    const int bh   = blockIdx.y;                    // b*H + h
    const int bidx = bh / HH;
    const int hidx = bh % HH;
    const int q0   = blockIdx.x * 64;
    const int qrow0 = q0 + wave * 16;

    // Q A-fragments for this wave's 16 rows (K = D = 64 -> 2 fragments)
    Frag aQ[2];
    {
        const __bf16* qp = qb + ((size_t)bh * LL + (qrow0 + l16)) * DD;
        #pragma unroll
        for (int kw = 0; kw < 2; ++kw) {
            aQ[kw].h[0] = *(const bf16x8*)(qp + kw * 32 + half * 8);
            aQ[kw].h[1] = *(const bf16x8*)(qp + kw * 32 + 16 + half * 8);
        }
    }

    v8f   accO[4] = {};
    float mSt[8], lSt[8];
    #pragma unroll
    for (int r = 0; r < 8; ++r) { mSt[r] = -3.0e38f; lSt[r] = 0.0f; }

    for (int kc = 0; kc < LL; kc += 64) {
        // ---- S = Q @ K^T (16x64 per wave), f32 accum -------------------
        v8f s[4];
        #pragma unroll
        for (int nt = 0; nt < 4; ++nt) {
            v8f c = {};
            const __bf16* kp = kb + ((size_t)bh * LL + (kc + nt * 16 + l16)) * DD;
            #pragma unroll
            for (int kw = 0; kw < 2; ++kw) {
                Frag bK;
                bK.h[0] = *(const bf16x8*)(kp + kw * 32 + half * 8);
                bK.h[1] = *(const bf16x8*)(kp + kw * 32 + 16 + half * 8);
                c = wmma_bf16(aQ[kw].v, bK.v, (v8f)c);
            }
            s[nt] = c;
        }

        // ---- mask (before scale, as in reference) ----------------------
        #pragma unroll
        for (int nt = 0; nt < 4; ++nt) {
            int col = kc + nt * 16 + l16;
            #pragma unroll
            for (int r = 0; r < 8; ++r) {
                int row = qrow0 + r + half * 8;
                int mv = mask[((size_t)bidx * LL + row) * LL + col];
                float e = s[nt][r];
                s[nt][r] = (mv == 0) ? -1.0e30f : e * SCALE;
            }
        }

        // ---- online softmax (rows live across 16 lanes of a half-wave) -
        #pragma unroll
        for (int r = 0; r < 8; ++r) {
            float vmax = fmaxf(fmaxf(s[0][r], s[1][r]), fmaxf(s[2][r], s[3][r]));
            vmax = redmax16(vmax);
            float mNew = fmaxf(mSt[r], vmax);
            float corr = __expf(mSt[r] - mNew);
            float rs = 0.0f;
            #pragma unroll
            for (int nt = 0; nt < 4; ++nt) {
                float pe = __expf(s[nt][r] - mNew);
                s[nt][r] = pe;
                rs += pe;
            }
            rs = redsum16(rs);
            lSt[r] = lSt[r] * corr + rs;
            mSt[r] = mNew;
            #pragma unroll
            for (int dt = 0; dt < 4; ++dt) accO[dt][r] *= corr;
        }

        // ---- relayout P: C-fragment -> A-fragment via LDS --------------
        __syncthreads();
        #pragma unroll
        for (int nt = 0; nt < 4; ++nt)
            #pragma unroll
            for (int r = 0; r < 8; ++r)
                sP[wave][r + half * 8][nt * 16 + l16] = (__bf16)s[nt][r];
        __syncthreads();

        Frag aP[2];
        #pragma unroll
        for (int kw = 0; kw < 2; ++kw) {
            const __bf16* pp = &sP[wave][l16][kw * 32 + half * 8];
            aP[kw].h[0] = *(const bf16x8*)(pp);
            aP[kw].h[1] = *(const bf16x8*)(pp + 16);
        }

        // ---- O += P @ V  (V stored [B,H,D,L] -> direct B-frag loads) ---
        #pragma unroll
        for (int dt = 0; dt < 4; ++dt) {
            const __bf16* vp = vb + ((size_t)bh * DD + dt * 16 + l16) * LL + kc;
            #pragma unroll
            for (int kw = 0; kw < 2; ++kw) {
                Frag bV;
                bV.h[0] = *(const bf16x8*)(vp + kw * 32 + half * 8);
                bV.h[1] = *(const bf16x8*)(vp + kw * 32 + 16 + half * 8);
                accO[dt] = wmma_bf16(aP[kw].v, bV.v, accO[dt]);
            }
        }
    }

    // ---- finalize: divide by row sums, write [B,L,E] f32 ---------------
    #pragma unroll
    for (int dt = 0; dt < 4; ++dt) {
        int d = dt * 16 + l16;
        #pragma unroll
        for (int r = 0; r < 8; ++r) {
            int row = qrow0 + r + half * 8;
            Ob[((size_t)bidx * LL + row) * EE + hidx * DD + d] = accO[dt][r] / lSt[r];
        }
    }
}

// ---------------------------------------------------------------------------
// Launcher
// ---------------------------------------------------------------------------
extern "C" void kernel_launch(void* const* d_in, const int* in_sizes, int n_in,
                              void* d_out, int out_size, void* d_ws, size_t ws_size,
                              hipStream_t stream) {
    (void)in_sizes; (void)n_in; (void)out_size; (void)ws_size;
    const float* values = (const float*)d_in[0];
    const float* keysI  = (const float*)d_in[1];
    const float* queryI = (const float*)d_in[2];
    const int*   mask   = (const int*)d_in[3];
    const float* Wv = (const float*)d_in[4];  const float* bv = (const float*)d_in[5];
    const float* Wk = (const float*)d_in[6];  const float* bk = (const float*)d_in[7];
    const float* Wq = (const float*)d_in[8];  const float* bq = (const float*)d_in[9];
    const float* Wo = (const float*)d_in[10]; const float* bo = (const float*)d_in[11];

    char* ws = (char*)d_ws;
    __bf16* qb = (__bf16*)(ws);                      // 4 MB  [B,H,L,D] bf16
    __bf16* kb = (__bf16*)(ws + (4u << 20));         // 4 MB  [B,H,L,D] bf16
    __bf16* vb = (__bf16*)(ws + (8u << 20));         // 4 MB  [B,H,D,L] bf16
    float*  Ob = (float*) (ws + (12u << 20));        // 8 MB  [B,L,E]   f32

    dim3 gGrid(GM / 128, EE / 128), gBlk(256);
    gemm_proj<0><<<gGrid, gBlk, 0, stream>>>(queryI, Wq, bq, (void*)qb);
    gemm_proj<0><<<gGrid, gBlk, 0, stream>>>(keysI,  Wk, bk, (void*)kb);
    gemm_proj<1><<<gGrid, gBlk, 0, stream>>>(values, Wv, bv, (void*)vb);

    attn_flash<<<dim3(LL / 64, BB * HH), 128, 0, stream>>>(qb, kb, vb, mask, Ob);

    gemm_proj<2><<<gGrid, gBlk, 0, stream>>>(Ob, Wo, bo, d_out);
}
